// SelfAttentionv1_89627377533630
// MI455X (gfx1250) — matrix operand order repaired and checked by reading, not compiled, not run on hardware
//
#include <hip/hip_runtime.h>
#include <hip/hip_bf16.h>

// Problem constants (match reference): B=4, S=4096, D=256
#define B_   4
#define S_   4096
#define D_   256
#define MTOT (B_ * S_)   // 16384 rows for the projection GEMMs

#define QB 128   // queries per workgroup (8 waves x 16 rows)
#define KB 32    // keys per staged block (one WMMA K=32 step for P@V)

typedef __attribute__((ext_vector_type(16))) __bf16   v16bf;
typedef __attribute__((ext_vector_type(8)))  __bf16   v8bf;
typedef __attribute__((ext_vector_type(8)))  float    v8f;
typedef __attribute__((ext_vector_type(4)))  uint32_t u32x4;
typedef __attribute__((ext_vector_type(8)))  uint32_t u32x8;

union Frag16 {
    v16bf v;
    v8bf  h[2];
    __bf16 e[16];
};

__device__ __forceinline__ v8f wmma_bf16(const Frag16& a, const Frag16& b, v8f c) {
    return __builtin_amdgcn_wmma_f32_16x16x32_bf16(
        false, a.v, false, b.v, (short)0, c, false, false);
}

// Wave-relative LDS byte offset of a __shared__ object (flat LDS address =
// {aperture_hi32, lds_offset32}; truncation gives the offset).
__device__ __forceinline__ uint32_t lds_off(const void* p) {
    return (uint32_t)(uintptr_t)p;
}

// Per-lane async DMA: LDS[lds_byte[lane]] <= MEM[g[lane]], 16B/lane, ASYNCcnt.
__device__ __forceinline__ void async_copy_b128(uint32_t lds_byte, const void* g) {
    asm volatile("global_load_async_to_lds_b128 %0, %1, off"
                 :: "v"(lds_byte), "v"(g) : "memory");
}
__device__ __forceinline__ void wait_async0() {
    asm volatile("s_wait_asynccnt 0x0" ::: "memory");
}

// Tensor Data Mover: DMA a 2D tile (tile0 x tile1, bf16) from a row-major
// tensor (inner dim tdim0, row stride stride0 elements) into LDS, packed.
// Descriptor bitfields per CDNA5 ISA. TENSORcnt.
__device__ __forceinline__ void tdm_load_2d(uint32_t lds_byte, const void* gptr,
                                            uint32_t tdim0, uint32_t tdim1,
                                            uint32_t tile0, uint32_t tile1,
                                            uint32_t stride0)
{
    uint64_t ga = (uint64_t)(uintptr_t)gptr;
    u32x4 g0 = { 1u,                                   // count=1 (valid D#)
                 lds_byte,                             // lds_addr
                 (uint32_t)ga,                         // global_addr[31:0]
                 (uint32_t)(ga >> 32) | (2u << 30) };  // addr[56:32] | type=2
    u32x8 g1 = { 0x00010000u,                          // data_size = 2 bytes
                 (tdim0 & 0xffffu) << 16,              // tensor_dim0 lo
                 (tdim0 >> 16) | ((tdim1 & 0xffffu) << 16),
                 (tdim1 >> 16) | (tile0 << 16),        // tile_dim0
                 tile1,                                // tile_dim1 | tile_dim2=0
                 stride0,                              // tensor_dim0_stride lo32
                 0u, 0u };
    u32x4 gz = { 0u, 0u, 0u, 0u };                     // dims 2+ unused
    asm volatile("tensor_load_to_lds %0, %1, %2, %3"
                 :: "s"(g0), "s"(g1), "s"(gz), "s"(gz) : "memory");
}

// ---------------------------------------------------------------------------
// Prep 1: x (f32) -> xbf (bf16), 8 elements / thread, pure b128 in / b128 out.
// ---------------------------------------------------------------------------
__global__ __launch_bounds__(256) void convert_x_kernel(
    const float* __restrict__ x, __bf16* __restrict__ xbf)
{
    const size_t i = ((size_t)blockIdx.x * 256 + threadIdx.x) * 8;
    float4 a = *(const float4*)(x + i);
    float4 b = *(const float4*)(x + i + 4);
    v8bf o;
    o[0] = (__bf16)a.x; o[1] = (__bf16)a.y; o[2] = (__bf16)a.z; o[3] = (__bf16)a.w;
    o[4] = (__bf16)b.x; o[5] = (__bf16)b.y; o[6] = (__bf16)b.z; o[7] = (__bf16)b.w;
    *(v8bf*)(xbf + i) = o;
}

// ---------------------------------------------------------------------------
// Prep 2: weights (f32 [k][n]) -> bf16 TRANSPOSED wT[n][k], so WMMA B-frags
// (lane = N, elements = consecutive K) are single 32-byte vector loads.
// grid = 4 matrices, 256 threads = one output row n each.
// ---------------------------------------------------------------------------
__global__ __launch_bounds__(256) void transpose_w_kernel(
    const float* __restrict__ wq, const float* __restrict__ wk,
    const float* __restrict__ wv, const float* __restrict__ wo,
    __bf16* __restrict__ wqT, __bf16* __restrict__ wkT,
    __bf16* __restrict__ wvT, __bf16* __restrict__ woT)
{
    const int m = blockIdx.x;
    const float* w = (m == 0) ? wq : (m == 1) ? wk : (m == 2) ? wv : wo;
    __bf16*     o = (m == 0) ? wqT : (m == 1) ? wkT : (m == 2) ? wvT : woT;
    const int n = threadIdx.x;
    for (int k = 0; k < D_; k += 8) {
        v8bf p;
        #pragma unroll
        for (int j = 0; j < 8; ++j) p[j] = (__bf16)w[(size_t)(k + j) * D_ + n];
        *(v8bf*)(o + (size_t)n * D_ + k) = p;
    }
}

// ---------------------------------------------------------------------------
// Kernel 1: fused QKV projection from bf16 x and bf16 transposed weights.
// Q/K row-major bf16; V stored TRANSPOSED as Vt[b][f][s] for TDM staging.
// A-frag: lane row = row0+lane%16, elem e -> K = k0+16*(lane/16)+e (contig, as B)
// B-frag: lane col = col0+lane%16, elem e -> K = k0+16*(lane/16)+e (contig!)
// ---------------------------------------------------------------------------
__global__ __launch_bounds__(256) void qkv_proj_kernel(
    const __bf16* __restrict__ xbf,
    const __bf16* __restrict__ wqT, const float* __restrict__ bq,
    const __bf16* __restrict__ wkT, const float* __restrict__ bk,
    const __bf16* __restrict__ wvT, const float* __restrict__ bv,
    __bf16* __restrict__ Q, __bf16* __restrict__ K, __bf16* __restrict__ Vt)
{
    const int tid   = threadIdx.x;
    const int wave  = tid >> 5;
    const int lane  = tid & 31;
    const int laneN = lane & 15;
    const int half  = lane >> 4;

    const int tilesPerMat = (MTOT / 16) * (D_ / 16); // 16384
    int tileId = blockIdx.x * 8 + wave;
    int mat    = tileId / tilesPerMat;
    int rem    = tileId % tilesPerMat;
    int row0   = (rem / (D_ / 16)) * 16;
    int col0   = (rem % (D_ / 16)) * 16;

    const __bf16* wT; const float* bias;
    if (mat == 0)      { wT = wqT; bias = bq; }
    else if (mat == 1) { wT = wkT; bias = bk; }
    else               { wT = wvT; bias = bv; }

    v8f acc = {};
    const __bf16* arow = xbf + (size_t)(row0 + laneN) * D_;
    const __bf16* brow = wT  + (size_t)(col0 + laneN) * D_;
    #pragma unroll
    for (int k0 = 0; k0 < D_; k0 += 32) {
        Frag16 a, b;
        const __bf16* ap = arow + k0 + 16 * half;
        a.h[0] = *(const v8bf*)ap;
        a.h[1] = *(const v8bf*)(ap + 8);
        b.v = *(const v16bf*)(brow + k0 + 16 * half);
        acc = wmma_bf16(a, b, acc);
    }

    const float bb = bias[col0 + laneN];
    if (mat < 2) {
        __bf16* out = (mat == 0) ? Q : K;
        #pragma unroll
        for (int v = 0; v < 8; ++v)
            out[(size_t)(row0 + v + 8 * half) * D_ + col0 + laneN] =
                (__bf16)(acc[v] + bb);
    } else {
        // Vt[b][f][s]: this lane's 8 rows are consecutive s -> one 16B store
        const int bidx = row0 / S_;
        const int s0   = (row0 % S_) + 8 * half;
        const int f    = col0 + laneN;
        v8bf pack;
        #pragma unroll
        for (int v = 0; v < 8; ++v) pack[v] = (__bf16)(acc[v] + bb);
        *(v8bf*)(Vt + ((size_t)bidx * D_ + f) * S_ + s0) = pack;
    }
}

// ---------------------------------------------------------------------------
// Kernel 2: causal flash attention.  K blocks staged with per-lane async DMA
// (ASYNCcnt), Vt blocks staged with the Tensor Data Mover (TENSORcnt), both
// double-buffered so DMA of block kb+1 overlaps the 32 WMMAs of block kb.
// ---------------------------------------------------------------------------
__global__ __launch_bounds__(256) void flash_attn_kernel(
    const __bf16* __restrict__ Q, const __bf16* __restrict__ K,
    const __bf16* __restrict__ Vt, __bf16* __restrict__ O)
{
    __shared__ __align__(128) __bf16 sK [2][KB * D_];   // [key][f]  2 x 16 KB
    __shared__ __align__(128) __bf16 sVt[2][D_ * KB];   // [f][key]  2 x 16 KB
    __shared__ __align__(128) __bf16 sP [8 * 16 * KB];  // per-wave P, 8 KB

    const int tid   = threadIdx.x;
    const int wave  = tid >> 5;
    const int lane  = tid & 31;
    const int laneN = lane & 15;
    const int half  = lane >> 4;

    const int b  = blockIdx.x / (S_ / QB);
    const int q0 = (blockIdx.x % (S_ / QB)) * QB;
    const int qw = q0 + wave * 16;

    const __bf16* Qb  = Q  + (size_t)b * S_ * D_;
    const __bf16* Kb  = K  + (size_t)b * S_ * D_;
    const __bf16* Vtb = Vt + (size_t)b * D_ * S_;

    // Preload Q fragments for this wave's 16 rows (16 x 256 bf16).
    Frag16 qa[8];
    #pragma unroll
    for (int kt = 0; kt < 8; ++kt) {
        const __bf16* qp = Qb + (size_t)(qw + laneN) * D_ + kt * 32 + 16 * half;
        qa[kt].h[0] = *(const v8bf*)qp;
        qa[kt].h[1] = *(const v8bf*)(qp + 8);
    }

    v8f acc[16];
    const v8f vzero = {};
    #pragma unroll
    for (int t = 0; t < 16; ++t) acc[t] = vzero;
    float m_i[8], l_i[8];
    #pragma unroll
    for (int v = 0; v < 8; ++v) { m_i[v] = -3.0e38f; l_i[v] = 0.0f; }

    __bf16* sPw = sP + wave * 16 * KB;
    const int nkb = (q0 + QB) / KB;  // causal upper bound for this block

    // DMA one 32-key block into buffer `buf` (no VGPR data path at all).
    auto stage = [&](int kb, int buf) {
        const int k0 = kb * KB;
        // K block: 32x256 bf16 contiguous -> 4 async b128 per thread
        #pragma unroll
        for (int j = 0; j < 4; ++j) {
            const int c = tid + j * 256;               // 16B chunk id
            async_copy_b128(lds_off(&sK[buf][(size_t)c * 8]),
                            Kb + (size_t)k0 * D_ + (size_t)c * 8);
        }
        // Vt block: 2D tile 32(s) x 256(f), row stride S_ -> one TDM descriptor
        if (wave == 0)
            tdm_load_2d(lds_off(&sVt[buf][0]), Vtb + k0,
                        /*tdim0=*/S_, /*tdim1=*/D_,
                        /*tile0=*/KB, /*tile1=*/D_, /*stride0=*/S_);
    };
    auto stage_wait = [&]() {
        wait_async0();
        if (wave == 0) __builtin_amdgcn_s_wait_tensorcnt((short)0);
    };

    stage(0, 0);
    stage_wait();
    __syncthreads();

    for (int kb = 0; kb < nkb; ++kb) {
        const int k0  = kb * KB;
        const int cur = kb & 1;

        if (kb + 1 < nkb) stage(kb + 1, cur ^ 1);      // overlap DMA w/ compute
        if (kb + 2 < nkb) {                            // warm L2 two ahead
            const char* nk = (const char*)(Kb + (size_t)(k0 + 2 * KB) * D_);
            __builtin_prefetch(nk + (size_t)(tid & 127) * 128, 0, 1);
            __builtin_prefetch(Vtb + (size_t)tid * S_ + (k0 + 2 * KB), 0, 1);
        }

        if (k0 <= qw + 15) {
            const __bf16* cK = sK [cur];
            const __bf16* cV = sVt[cur];

            // ---- S = Q . K^T over the 32-key block ------------------------
            v8f s0 = vzero, s1 = vzero;
            #pragma unroll
            for (int kt = 0; kt < 8; ++kt) {
                Frag16 b0, b1;
                b0.v = *(const v16bf*)(cK + (size_t)laneN        * D_ + kt * 32 + 16 * half);
                b1.v = *(const v16bf*)(cK + (size_t)(16 + laneN) * D_ + kt * 32 + 16 * half);
                s0 = wmma_bf16(qa[kt], b0, s0);
                s1 = wmma_bf16(qa[kt], b1, s1);
            }

            // ---- online softmax ------------------------------------------
            #pragma unroll
            for (int v = 0; v < 8; ++v) {
                const int row = qw + v + 8 * half;
                float e0 = s0[v] * 0.0625f;                 // 1/sqrt(256)
                float e1 = s1[v] * 0.0625f;
                if (k0 + laneN > row)      e0 = -1.0e9f;    // causal NEG
                if (k0 + 16 + laneN > row) e1 = -1.0e9f;

                float mx = fmaxf(e0, e1);
                #pragma unroll
                for (int off = 8; off >= 1; off >>= 1)
                    mx = fmaxf(mx, __shfl_xor(mx, off, 32));
                const float mnew  = fmaxf(m_i[v], mx);
                const float alpha = __expf(m_i[v] - mnew);
                const float p0 = __expf(e0 - mnew);
                const float p1 = __expf(e1 - mnew);
                float rs = p0 + p1;
                #pragma unroll
                for (int off = 8; off >= 1; off >>= 1)
                    rs += __shfl_xor(rs, off, 32);
                l_i[v] = l_i[v] * alpha + rs;
                m_i[v] = mnew;
                #pragma unroll
                for (int t = 0; t < 16; ++t) acc[t][v] *= alpha;

                const int prow = v + 8 * half;
                sPw[prow * KB + laneN]      = (__bf16)p0;
                sPw[prow * KB + 16 + laneN] = (__bf16)p1;
            }
            asm volatile("s_wait_dscnt 0x0" ::: "memory");  // per-wave P RAW

            // ---- O += P . V ----------------------------------------------
            Frag16 pa;
            const __bf16* pp = sPw + laneN * KB + 16 * half;
            pa.h[0] = *(const v8bf*)pp;
            pa.h[1] = *(const v8bf*)(pp + 8);
            #pragma unroll
            for (int t = 0; t < 16; ++t) {
                Frag16 bf;
                bf.v = *(const v16bf*)(cV + (size_t)(t * 16 + laneN) * KB + 16 * half);
                acc[t] = wmma_bf16(pa, bf, acc[t]);
            }
        }

        stage_wait();        // next buffer complete before anyone advances
        __syncthreads();
    }

    // ---- normalize and emit attention output (bf16) -------------------------
    __bf16* Ob = O + (size_t)b * S_ * D_;
    #pragma unroll
    for (int v = 0; v < 8; ++v) {
        const float inv = 1.0f / l_i[v];
        const int row = qw + v + 8 * half;
        #pragma unroll
        for (int t = 0; t < 16; ++t)
            Ob[(size_t)row * D_ + t * 16 + laneN] = (__bf16)(acc[t][v] * inv);
    }
}

// ---------------------------------------------------------------------------
// Kernel 3: output projection  out = score(bf16) @ woT^T + bo   (fp32 out)
// ---------------------------------------------------------------------------
__global__ __launch_bounds__(256) void out_proj_kernel(
    const __bf16* __restrict__ SC, const __bf16* __restrict__ woT,
    const float* __restrict__ bo, float* __restrict__ out)
{
    const int tid   = threadIdx.x;
    const int wave  = tid >> 5;
    const int lane  = tid & 31;
    const int laneN = lane & 15;
    const int half  = lane >> 4;

    int tileId = blockIdx.x * 8 + wave;
    int row0   = (tileId / (D_ / 16)) * 16;
    int col0   = (tileId % (D_ / 16)) * 16;

    v8f acc = {};
    const __bf16* arow = SC  + (size_t)(row0 + laneN) * D_;
    const __bf16* brow = woT + (size_t)(col0 + laneN) * D_;
    #pragma unroll
    for (int k0 = 0; k0 < D_; k0 += 32) {
        Frag16 a, b;
        const __bf16* ap = arow + k0 + 16 * half;
        a.h[0] = *(const v8bf*)ap;
        a.h[1] = *(const v8bf*)(ap + 8);
        b.v = *(const v16bf*)(brow + k0 + 16 * half);
        acc = wmma_bf16(a, b, acc);
    }

    const float bb = bo[col0 + laneN];
    #pragma unroll
    for (int v = 0; v < 8; ++v)
        out[(size_t)(row0 + v + 8 * half) * D_ + col0 + laneN] = acc[v] + bb;
}

// ---------------------------------------------------------------------------
extern "C" void kernel_launch(void* const* d_in, const int* in_sizes, int n_in,
                              void* d_out, int out_size, void* d_ws, size_t ws_size,
                              hipStream_t stream)
{
    const float* x  = (const float*)d_in[0];
    // d_in[1]: causal tril mask, applied analytically in-kernel.
    const float* wq = (const float*)d_in[2];
    const float* bq = (const float*)d_in[3];
    const float* wk = (const float*)d_in[4];
    const float* bk = (const float*)d_in[5];
    const float* wv = (const float*)d_in[6];
    const float* bv = (const float*)d_in[7];
    const float* wo = (const float*)d_in[8];
    const float* bo = (const float*)d_in[9];
    float* out = (float*)d_out;

    const size_t matElems = (size_t)MTOT * D_;   // 4,194,304
    __bf16* Q   = (__bf16*)d_ws;
    __bf16* K   = Q   + matElems;
    __bf16* Vt  = K   + matElems;   // transposed: [b][f][s]
    __bf16* SC  = Vt  + matElems;
    __bf16* xbf = SC  + matElems;
    __bf16* wqT = xbf + matElems;
    __bf16* wkT = wqT + (size_t)D_ * D_;
    __bf16* wvT = wkT + (size_t)D_ * D_;
    __bf16* woT = wvT + (size_t)D_ * D_;

    convert_x_kernel<<<matElems / (256 * 8), 256, 0, stream>>>(x, xbf);
    transpose_w_kernel<<<4, 256, 0, stream>>>(wq, wk, wv, wo, wqT, wkT, wvT, woT);

    qkv_proj_kernel<<<3 * (MTOT / 16) * (D_ / 16) / 8, 256, 0, stream>>>(
        xbf, wqT, bq, wkT, bk, wvT, bv, Q, K, Vt);

    flash_attn_kernel<<<B_ * (S_ / QB), 256, 0, stream>>>(Q, K, Vt, SC);

    out_proj_kernel<<<(MTOT / 16) * (D_ / 16) / 8, 256, 0, stream>>>(
        SC, woT, bo, out);
}
